// GCL_38431367365234
// MI455X (gfx1250) — matrix operand order repaired and checked by reading, not compiled, not run on hardware
//
#include <hip/hip_runtime.h>
#include <hip/hip_bf16.h>

// ---------------- problem constants (match reference) ----------------
#define N_NODES 20000
#define T_TYPES 4
#define H_DIM   256
#define E_EDGES 160000
#define EK_EDGES 40000
#define HEADS 32
#define DHD 8

typedef __attribute__((ext_vector_type(16))) _Float16 v16h;
typedef __attribute__((ext_vector_type(8)))  float    v8f;

// silu via hardware transcendentals: v_exp_f32 + v_rcp_f32 (1-ulp), avoids the
// ~10-instruction IEEE divide expansion per element.
__device__ __forceinline__ float silu_f(float x) {
    return x * __builtin_amdgcn_rcpf(1.0f + __expf(-x));
}

// D = A(16x32 f16) * B(32x16 f16) + C (f32), wave32 WMMA
__device__ __forceinline__ v8f wmma_f16(v16h a, v16h b, v8f c) {
    return __builtin_amdgcn_wmma_f32_16x16x32_f16(
        /*neg_a=*/false, a, /*neg_b=*/false, b,
        /*c_mod=*/(short)0, c, /*reuse_a=*/false, /*reuse_b=*/false);
}

// A fragment: row-major [16][lda] (LDS), tile rows m = lane&15.
// ISA 16-bit A 16x32 layout: lanes 0-15 halves 0..7 -> K 0..7, halves 8..15 -> K 16..23;
// lanes 16-31 shifted by +8 in K. Both 8-half groups are k-contiguous so the
// compiler lowers this to two ds_load_b128.
__device__ __forceinline__ v16h load_frag_a(const _Float16* __restrict__ A,
                                            int lda, int k0, int lane) {
    v16h a;
    const int m    = lane & 15;
    const int koff = (lane >> 4) << 3;
    #pragma unroll
    for (int i = 0; i < 16; ++i) {
        const int k = (i & 7) + ((i >> 3) << 4) + koff;
        a[i] = A[m * lda + k0 + k];
    }
    return a;
}

// Packed-B fragment: weights pre-swizzled into fragment-major order at
// conversion time. Fragment `frag` occupies 512 contiguous halves laid out
// [lane][16 halves]; per-lane load is one aligned 32-byte vector
// (-> two global_load_b128).
__device__ __forceinline__ v16h load_frag_b_packed(const _Float16* __restrict__ Bp,
                                                   int frag, int lane) {
    return *(const v16h*)(Bp + ((size_t)frag << 9) + ((size_t)lane << 4));
}

// ---------------- prep kernels ----------------
// Pack f32 weight matrices [K][Nout] (nmat of them, contiguous) into f16
// fragment-major layout: dst[frag*512 + L*16 + i] =
//   src[(kt*32 + ((L>>4)<<3) + (i&7) + ((i>>3)<<4)) * Nout + nt*16 + (L&15)],  frag = kt*ntiles+nt
__global__ void pack_b_kernel(const float* __restrict__ src,
                              _Float16* __restrict__ dst,
                              int K, int Nout, int nmat) {
    const long long per = (long long)K * Nout;
    long long p = (long long)blockIdx.x * blockDim.x + threadIdx.x;
    if (p >= per * nmat) return;
    const int mat = (int)(p / per);
    const int q   = (int)(p % per);
    const int ntiles = Nout >> 4;
    const int frag = q >> 9;
    const int r    = q & 511;
    const int L    = r >> 4;
    const int i    = r & 15;
    const int kt   = frag / ntiles;
    const int nt   = frag % ntiles;
    const int k = kt * 32 + ((L >> 4) << 3) + (i & 7) + ((i >> 3) << 4);
    const int n = nt * 16 + (L & 15);
    dst[mat * per + q] = (_Float16)src[mat * per + (size_t)k * Nout + n];
}

__global__ void zero_f32_kernel(float* __restrict__ p, long long n) {
    long long i = (long long)blockIdx.x * blockDim.x + threadIdx.x;
    if (i < n) p[i] = 0.0f;
}

// node_emb[n,f] = sum_t h[n,t,f]*nc_w[t] + nc_b   -> f16
__global__ void node_emb_kernel(const float* __restrict__ h,
                                const float* __restrict__ nc_w,
                                const float* __restrict__ nc_b,
                                _Float16* __restrict__ ne16) {
    long long i = (long long)blockIdx.x * blockDim.x + threadIdx.x;
    if (i >= (long long)N_NODES * H_DIM) return;
    const int f = (int)(i & (H_DIM - 1));
    const long long n = i >> 8;
    const float* hp = h + (n * T_TYPES) * H_DIM + f;
    float acc = nc_b[0];
    #pragma unroll
    for (int t = 0; t < T_TYPES; ++t) acc += hp[(size_t)t * H_DIM] * nc_w[t];
    ne16[i] = (_Float16)acc;
}

// ---------------- fused edge MLP: 16 edges per block ----------------
// x = concat(src,tgt) [16,512]; silu(x@w1+b1) [16,768]; silu(.@w2+b2) [16,512];
// mij = .@w3+b3 [16,256]  -> fp32 to d_out's mij section
__global__ __launch_bounds__(256) void edge_mlp_kernel(
    const _Float16* __restrict__ ne16,
    const int* __restrict__ edge_index,
    const _Float16* __restrict__ w1p, const float* __restrict__ b1,
    const _Float16* __restrict__ w2p, const float* __restrict__ b2,
    const _Float16* __restrict__ w3p, const float* __restrict__ b3,
    float* __restrict__ mij_out) {
    __shared__ _Float16 a0[16][512];   // 16 KB
    __shared__ _Float16 o1[16][768];   // 24 KB
    __shared__ _Float16 o2[16][512];   // 16 KB
    const int tid = threadIdx.x;
    const int lane = tid & 31;
    const int wave = tid >> 5;
    const int eb = blockIdx.x * 16;

    // gather concat(src_emb, tgt_emb) into LDS
    for (int idx = tid; idx < 16 * 512; idx += 256) {
        const int r = idx >> 9, c = idx & 511;
        const int e = eb + r;
        const int node = (c < H_DIM) ? edge_index[e] : edge_index[E_EDGES + e];
        a0[r][c] = ne16[(size_t)node * H_DIM + (c & (H_DIM - 1))];
    }
    __syncthreads();

    // layer1: [16,512]x[512,768], 48 N-tiles across 8 waves
    for (int nt = wave; nt < 48; nt += 8) {
        const int n0 = nt * 16;
        v8f c = {};
        for (int kt = 0; kt < 16; ++kt) {
            const int frag = kt * 48 + nt;
            if (kt + 1 < 16)
                __builtin_prefetch(w1p + ((size_t)(frag + 48) << 9) + ((size_t)lane << 4), 0, 3);
            v16h a = load_frag_a(&a0[0][0], 512, kt * 32, lane);
            v16h b = load_frag_b_packed(w1p, frag, lane);
            c = wmma_f16(a, b, c);
        }
        const int n = lane & 15;
        const float bias = b1[n0 + n];
        #pragma unroll
        for (int j = 0; j < 8; ++j) {
            const int m = ((lane >> 4) << 3) + j;
            o1[m][n0 + n] = (_Float16)silu_f(c[j] + bias);
        }
    }
    __syncthreads();

    // layer2: [16,768]x[768,512], 32 N-tiles
    for (int nt = wave; nt < 32; nt += 8) {
        const int n0 = nt * 16;
        v8f c = {};
        for (int kt = 0; kt < 24; ++kt) {
            v16h a = load_frag_a(&o1[0][0], 768, kt * 32, lane);
            v16h b = load_frag_b_packed(w2p, kt * 32 + nt, lane);
            c = wmma_f16(a, b, c);
        }
        const int n = lane & 15;
        const float bias = b2[n0 + n];
        #pragma unroll
        for (int j = 0; j < 8; ++j) {
            const int m = ((lane >> 4) << 3) + j;
            o2[m][n0 + n] = (_Float16)silu_f(c[j] + bias);
        }
    }
    __syncthreads();

    // layer3: [16,512]x[512,256] -> mij fp32 (no activation)
    for (int nt = wave; nt < 16; nt += 8) {
        const int n0 = nt * 16;
        v8f c = {};
        for (int kt = 0; kt < 16; ++kt) {
            v16h a = load_frag_a(&o2[0][0], 512, kt * 32, lane);
            v16h b = load_frag_b_packed(w3p, kt * 16 + nt, lane);
            c = wmma_f16(a, b, c);
        }
        const int n = lane & 15;
        const float bias = b3[n0 + n];
        #pragma unroll
        for (int j = 0; j < 8; ++j) {
            const int m = ((lane >> 4) << 3) + j;
            mij_out[(size_t)(eb + m) * H_DIM + n0 + n] = c[j] + bias;
        }
    }
}

// ---------------- GINE message + scatter-add ----------------
// one thread handles one (t, ek, 4-float chunk)
__global__ __launch_bounds__(256) void gine_message_kernel(
    const float* __restrict__ h,
    const int* __restrict__ edge_index,
    const int* __restrict__ edge_mask,
    const float* __restrict__ mij,     // [E,H] fp32 (in d_out)
    float* __restrict__ agg) {         // [T,N,H] fp32
    const long long gid = (long long)blockIdx.x * blockDim.x + threadIdx.x;
    const int FCH = H_DIM / 4;         // 64 chunks per edge
    const int fi = (int)(gid % FCH) * 4;
    const long long te = gid / FCH;
    const int ek = (int)(te % EK_EDGES);
    const int t  = (int)(te / EK_EDGES);
    if (t >= T_TYPES) return;
    const int m   = edge_mask[t * EK_EDGES + ek];
    const int row = edge_index[m];
    const int col = edge_index[E_EDGES + m];
    const float4 hv = *(const float4*)&h[((size_t)row * T_TYPES + t) * H_DIM + fi];
    const float4 ev = *(const float4*)&mij[(size_t)m * H_DIM + fi];
    float* dst = &agg[(((size_t)t * N_NODES) + col) * H_DIM + fi];
    atomicAdd(dst + 0, silu_f(hv.x + ev.x));
    atomicAdd(dst + 1, silu_f(hv.y + ev.y));
    atomicAdd(dst + 2, silu_f(hv.z + ev.z));
    atomicAdd(dst + 3, silu_f(hv.w + ev.w));
}

// ---------------- GINE residual MLP (4 layers) ----------------
// one block = 16 nodes x 1 type; z kept in LDS (f32 residual + f16 WMMA copy)
__global__ __launch_bounds__(256) void conv_mlp_kernel(
    const float* __restrict__ h,
    const float* __restrict__ agg,
    const _Float16* __restrict__ cwp,    // [T,4] packed [256,256] f16
    const float* __restrict__ cb,        // [T,4,H] f32
    _Float16* __restrict__ hid16) {      // [N,T,H] f16 out
    __shared__ float    zf[16][256];     // 16 KB
    __shared__ _Float16 zh[16][256];     //  8 KB
    __shared__ float    yf[16][256];     // 16 KB
    const int tid = threadIdx.x, lane = tid & 31, wave = tid >> 5;
    const int t  = blockIdx.y;
    const int n0 = blockIdx.x * 16;

    for (int idx = tid; idx < 16 * 256; idx += 256) {
        const int r = idx >> 8, c = idx & 255;
        const int n = n0 + r;
        const float z = h[((size_t)n * T_TYPES + t) * H_DIM + c] +
                        agg[(((size_t)t * N_NODES) + n) * H_DIM + c];
        zf[r][c] = z;
        zh[r][c] = (_Float16)z;
    }
    __syncthreads();

    for (int l = 0; l < 4; ++l) {
        const _Float16* Bp = cwp + (size_t)(t * 4 + l) * H_DIM * H_DIM;
        const float* bias  = cb + (size_t)(t * 4 + l) * H_DIM;
        for (int nt = wave; nt < 16; nt += 8) {
            const int c0 = nt * 16;
            v8f c = {};
            for (int kt = 0; kt < 8; ++kt) {
                v16h a = load_frag_a(&zh[0][0], 256, kt * 32, lane);
                v16h b = load_frag_b_packed(Bp, kt * 16 + nt, lane);
                c = wmma_f16(a, b, c);
            }
            const int n = lane & 15;
            const float bv = bias[c0 + n];
            #pragma unroll
            for (int j = 0; j < 8; ++j) {
                const int m = ((lane >> 4) << 3) + j;
                float y = c[j] + bv;
                if (l < 3) y = silu_f(y);
                yf[m][c0 + n] = y;
            }
        }
        __syncthreads();
        for (int idx = tid; idx < 16 * 256; idx += 256) {
            const int r = idx >> 8, c = idx & 255;
            const float z = yf[r][c] + zf[r][c];   // residual
            zf[r][c] = z;
            zh[r][c] = (_Float16)z;
        }
        __syncthreads();
    }

    // hid = silu(z) + xb, stored in [N,T,H] layout for attention
    for (int idx = tid; idx < 16 * 256; idx += 256) {
        const int r = idx >> 8, c = idx & 255;
        const int n = n0 + r;
        const float v = silu_f(zf[r][c]) + h[((size_t)n * T_TYPES + t) * H_DIM + c];
        hid16[((size_t)n * T_TYPES + t) * H_DIM + c] = (_Float16)v;
    }
}

// ---------------- fused cross-branch attention ----------------
// one block = 8 nodes (32 rows of [N*T, H]); q/k/v via WMMA; 4x4 softmax per
// (node, head); o@wo + bo + residual via WMMA -> d_out
__global__ __launch_bounds__(256) void attn_kernel(
    const _Float16* __restrict__ hid16,
    const _Float16* __restrict__ wqp, const float* __restrict__ bq,
    const _Float16* __restrict__ wkp, const float* __restrict__ bk,
    const _Float16* __restrict__ wvp, const float* __restrict__ bv,
    const _Float16* __restrict__ wop, const float* __restrict__ bo,
    float* __restrict__ out) {
    __shared__ _Float16 Ax[32][256];   // 16 KB input rows (also residual)
    __shared__ _Float16 Qs[32][256];   // 16 KB q (reused for o)
    __shared__ _Float16 Ks[32][256];   // 16 KB k
    __shared__ _Float16 Vs[32][256];   // 16 KB v
    const int tid = threadIdx.x, lane = tid & 31, wave = tid >> 5;
    const size_t r0 = (size_t)blockIdx.x * 32;

    for (int idx = tid; idx < 32 * 256; idx += 256) {
        const int r = idx >> 8, c = idx & 255;
        Ax[r][c] = hid16[(r0 + r) * H_DIM + c];
    }
    __syncthreads();

    // q,k,v projections: 2 Mtiles x 16 Ntiles x 3 mats = 96 tiles over 8 waves
    for (int tg = wave; tg < 96; tg += 8) {
        const int mat  = tg / 32;
        const int tile = tg % 32;
        const int mt = tile >> 4, nt = tile & 15;
        const int c0 = nt * 16;
        const _Float16* Wp = (mat == 0) ? wqp : (mat == 1) ? wkp : wvp;
        const float* bias  = (mat == 0) ? bq  : (mat == 1) ? bk  : bv;
        v8f c = {};
        for (int kt = 0; kt < 8; ++kt) {
            v16h a = load_frag_a(&Ax[mt * 16][0], 256, kt * 32, lane);
            v16h b = load_frag_b_packed(Wp, kt * 16 + nt, lane);
            c = wmma_f16(a, b, c);
        }
        const int n = lane & 15;
        const float bvv = bias[c0 + n];
        _Float16 (*Dst)[256] = (mat == 0) ? Qs : (mat == 1) ? Ks : Vs;
        #pragma unroll
        for (int j = 0; j < 8; ++j) {
            const int m = mt * 16 + ((lane >> 4) << 3) + j;
            Dst[m][c0 + n] = (_Float16)(c[j] + bvv);
        }
    }
    __syncthreads();

    // per-(node, head) softmax over the 4 type tokens; 256 pairs = 1/thread
    const int nd = tid >> 5;   // node in block: 0..7
    const int hd = tid & 31;   // head: 0..31
    float o[4][DHD];
    {
        const float scale = 0.35355339059327373f;  // 1/sqrt(8)
        float s[4][4];
        #pragma unroll
        for (int i = 0; i < 4; ++i)
            #pragma unroll
            for (int j = 0; j < 4; ++j) {
                float acc = 0.0f;
                #pragma unroll
                for (int d = 0; d < DHD; ++d)
                    acc += (float)Qs[nd * 4 + i][hd * DHD + d] *
                           (float)Ks[nd * 4 + j][hd * DHD + d];
                s[i][j] = acc * scale;
            }
        #pragma unroll
        for (int i = 0; i < 4; ++i) {
            const float mx = fmaxf(fmaxf(s[i][0], s[i][1]), fmaxf(s[i][2], s[i][3]));
            const float e0 = __expf(s[i][0] - mx), e1 = __expf(s[i][1] - mx);
            const float e2 = __expf(s[i][2] - mx), e3 = __expf(s[i][3] - mx);
            const float inv = __builtin_amdgcn_rcpf(e0 + e1 + e2 + e3);
            #pragma unroll
            for (int d = 0; d < DHD; ++d) {
                o[i][d] = inv * (e0 * (float)Vs[nd * 4 + 0][hd * DHD + d] +
                                 e1 * (float)Vs[nd * 4 + 1][hd * DHD + d] +
                                 e2 * (float)Vs[nd * 4 + 2][hd * DHD + d] +
                                 e3 * (float)Vs[nd * 4 + 3][hd * DHD + d]);
            }
        }
    }
    __syncthreads();
    #pragma unroll
    for (int i = 0; i < 4; ++i)
        #pragma unroll
        for (int d = 0; d < DHD; ++d)
            Qs[nd * 4 + i][hd * DHD + d] = (_Float16)o[i][d];   // reuse Qs as o
    __syncthreads();

    // out = o @ wo + bo + xa (residual), 32 tiles over 8 waves
    for (int tile = wave; tile < 32; tile += 8) {
        const int mt = tile >> 4, nt = tile & 15;
        const int c0 = nt * 16;
        v8f c = {};
        for (int kt = 0; kt < 8; ++kt) {
            v16h a = load_frag_a(&Qs[mt * 16][0], 256, kt * 32, lane);
            v16h b = load_frag_b_packed(wop, kt * 16 + nt, lane);
            c = wmma_f16(a, b, c);
        }
        const int n = lane & 15;
        const float bvv = bo[c0 + n];
        #pragma unroll
        for (int j = 0; j < 8; ++j) {
            const int m = mt * 16 + ((lane >> 4) << 3) + j;
            out[(r0 + m) * H_DIM + c0 + n] = c[j] + bvv + (float)Ax[m][c0 + n];
        }
    }
}

// ---------------- host launcher ----------------
extern "C" void kernel_launch(void* const* d_in, const int* in_sizes, int n_in,
                              void* d_out, int out_size, void* d_ws, size_t ws_size,
                              hipStream_t stream) {
    const float* h          = (const float*)d_in[0];
    const int*   edge_index = (const int*)d_in[1];
    const int*   edge_mask  = (const int*)d_in[2];
    const float* nc_w       = (const float*)d_in[3];
    const float* nc_b       = (const float*)d_in[4];
    const float* em_w1      = (const float*)d_in[5];
    const float* em_b1      = (const float*)d_in[6];
    const float* em_w2      = (const float*)d_in[7];
    const float* em_b2      = (const float*)d_in[8];
    const float* em_w3      = (const float*)d_in[9];
    const float* em_b3      = (const float*)d_in[10];
    const float* conv_w     = (const float*)d_in[11];
    const float* conv_b     = (const float*)d_in[12];
    const float* wq         = (const float*)d_in[13];
    const float* bq         = (const float*)d_in[14];
    const float* wk         = (const float*)d_in[15];
    const float* bk         = (const float*)d_in[16];
    const float* wv         = (const float*)d_in[17];
    const float* bv         = (const float*)d_in[18];
    const float* wo         = (const float*)d_in[19];
    const float* bo         = (const float*)d_in[20];

    float* out = (float*)d_out;                                   // [N,T,H]
    float* mij = out + (size_t)N_NODES * T_TYPES * H_DIM;         // [E,H]

    // workspace carve (fixed deterministic layout)
    char* ws = (char*)d_ws;
    size_t off = 0;
    auto carve = [&](size_t bytes) -> char* {
        char* p = ws + off;
        off += (bytes + 255) & ~(size_t)255;
        return p;
    };
    _Float16* ne16  = (_Float16*)carve((size_t)N_NODES * H_DIM * 2);
    _Float16* w1p   = (_Float16*)carve((size_t)512 * 768 * 2);
    _Float16* w2p   = (_Float16*)carve((size_t)768 * 512 * 2);
    _Float16* w3p   = (_Float16*)carve((size_t)512 * 256 * 2);
    _Float16* cwp   = (_Float16*)carve((size_t)T_TYPES * 4 * H_DIM * H_DIM * 2);
    _Float16* wqp   = (_Float16*)carve((size_t)H_DIM * H_DIM * 2);
    _Float16* wkp   = (_Float16*)carve((size_t)H_DIM * H_DIM * 2);
    _Float16* wvp   = (_Float16*)carve((size_t)H_DIM * H_DIM * 2);
    _Float16* wop   = (_Float16*)carve((size_t)H_DIM * H_DIM * 2);
    float*    agg   = (float*)carve((size_t)T_TYPES * N_NODES * H_DIM * 4);
    _Float16* hid16 = (_Float16*)carve((size_t)N_NODES * T_TYPES * H_DIM * 2);

    auto pack = [&](const float* s, _Float16* d, int K, int Nout, int nmat) {
        const long long n = (long long)K * Nout * nmat;
        pack_b_kernel<<<(int)((n + 255) / 256), 256, 0, stream>>>(s, d, K, Nout, nmat);
    };
    pack(em_w1, w1p, 512, 768, 1);
    pack(em_w2, w2p, 768, 512, 1);
    pack(em_w3, w3p, 512, 256, 1);
    pack(conv_w, cwp, 256, 256, 16);   // [T,4] H x H matrices
    pack(wq, wqp, 256, 256, 1);
    pack(wk, wkp, 256, 256, 1);
    pack(wv, wvp, 256, 256, 1);
    pack(wo, wop, 256, 256, 1);

    {
        const long long n = (long long)N_NODES * H_DIM;
        node_emb_kernel<<<(int)((n + 255) / 256), 256, 0, stream>>>(h, nc_w, nc_b, ne16);
    }
    {
        const long long n = (long long)T_TYPES * N_NODES * H_DIM;
        zero_f32_kernel<<<(int)((n + 255) / 256), 256, 0, stream>>>(agg, n);
    }

    // edge MLP -> mij (fp32, d_out section 2)
    edge_mlp_kernel<<<E_EDGES / 16, 256, 0, stream>>>(
        ne16, edge_index, w1p, em_b1, w2p, em_b2, w3p, em_b3, mij);

    // message + scatter-add into agg
    {
        const long long n = (long long)T_TYPES * EK_EDGES * (H_DIM / 4);
        gine_message_kernel<<<(int)((n + 255) / 256), 256, 0, stream>>>(
            h, edge_index, edge_mask, mij, agg);
    }

    // residual conv MLP -> hid16 [N,T,H]
    conv_mlp_kernel<<<dim3(N_NODES / 16, T_TYPES), 256, 0, stream>>>(
        h, agg, cwp, conv_b, hid16);

    // fused attention -> out
    attn_kernel<<<(N_NODES * T_TYPES) / 32, 256, 0, stream>>>(
        hid16, wqp, bq, wkp, bk, wvp, bv, wop, bo, out);

    (void)in_sizes; (void)n_in; (void)out_size; (void)ws_size;
}
